// EfficientTransformerLayer_20624432956129
// MI455X (gfx1250) — matrix operand order repaired and checked by reading, not compiled, not run on hardware
//
#include <hip/hip_runtime.h>
#include <math.h>

// ---------------- model constants ----------------
#define D_MODEL 1024
#define NHEAD   16
#define HDIM    64
#define SEQ     2048
#define NBATCH  2
#define MTOK    (NBATCH * SEQ)   // 4096 tokens
#define ATT_SCALE 0.125f         // 64^-0.5

typedef __bf16 bf16_t;
typedef __attribute__((ext_vector_type(16))) __bf16        v16bf;
typedef __attribute__((ext_vector_type(8)))  float         v8f;
typedef __attribute__((ext_vector_type(4)))  unsigned int  u32x4;

union Frag { u32x4 q[2]; v16bf v; };

static __device__ __forceinline__ v8f wmma_bf16(v16bf a, v16bf b, v8f c) {
  // D = A(16x32 bf16) * B(32x16 bf16) + C(16x16 f32)
  return __builtin_amdgcn_wmma_f32_16x16x32_bf16(false, a, false, b, (short)0, c, false, false);
}

// butterfly-xor via ds_swizzle immediate (and_mask=0x1f, or=0, xor=M) — no address VGPR
template <int M>
static __device__ __forceinline__ float swz_xor(float x) {
  return __int_as_float(__builtin_amdgcn_ds_swizzle(__float_as_int(x), (M << 10) | 0x1f));
}

// ---- CDNA5 async Global->LDS DMA (no VGPR data round-trip, tracked by ASYNCcnt) ----
// GV mode: per-lane LDS byte offset in VDST VGPR, 64-bit global address in VGPR pair.
static __device__ __forceinline__ void async_ld_b128(void* lds, const void* gaddr) {
  unsigned l = (unsigned)(size_t)lds;               // low 32b of generic ptr = LDS offset
  unsigned long long g = (unsigned long long)(size_t)gaddr;
  asm volatile("global_load_async_to_lds_b128 %0, %1, off"
               :: "v"(l), "v"(g) : "memory");
}
template <int N>
static __device__ __forceinline__ void wait_asynccnt() {
  asm volatile("s_wait_asynccnt %0" :: "n"(N) : "memory");
}

// ---------------- weight transpose + fp32->bf16 ----------------
// in: [K, N] fp32 row-major ; out: [N, K] bf16 row-major
__global__ __launch_bounds__(256) void k_transpose_cvt(const float* __restrict__ in,
                                                       bf16_t* __restrict__ out,
                                                       int K, int N) {
  __shared__ float tile[32][33];
  int nb = blockIdx.x * 32, kb = blockIdx.y * 32;
  int tx = threadIdx.x, ty = threadIdx.y;           // 32 x 8
#pragma unroll
  for (int i = 0; i < 32; i += 8)
    tile[ty + i][tx] = in[(size_t)(kb + ty + i) * N + (nb + tx)];
  __syncthreads();
#pragma unroll
  for (int i = 0; i < 32; i += 8)
    out[(size_t)(nb + ty + i) * K + (kb + tx)] = (bf16_t)tile[tx][ty + i];
}

// ---------------- elementwise fp32 -> bf16 ----------------
__global__ __launch_bounds__(256) void k_cvt_bf16(const float* __restrict__ in,
                                                  bf16_t* __restrict__ out, int n) {
  int i = blockIdx.x * 256 + threadIdx.x;
  if (i < n) out[i] = (bf16_t)in[i];
}

// ---------------- generic bf16 GEMM: C[M,N] = A[M,K] * Bt[N,K]^T ----------------
// block tile 128x128x32, 256 threads = 8 waves (2x4 wmma tiles each).
// Double-buffered LDS fed by async Global->LDS DMA: tile k+1 streams in
// while tile k's WMMAs execute; s_wait_asynccnt 4 retires the older tile.
#define BM 128
#define BN 128
#define BK 32
#define LDA 48   // padded row (bf16 elems), 96B: 16B aligned, spread banks
#define LDB 48

enum { EPI_QKV = 0, EPI_RES = 1, EPI_GELU = 2, EPI_FFN2 = 3 };

template <int EPI>
__global__ __launch_bounds__(256)
void k_gemm_bf16(const bf16_t* __restrict__ A, const bf16_t* __restrict__ Bt,
                 int M, int N, int K,
                 const float* __restrict__ bias, const float* __restrict__ res,
                 float* __restrict__ outF,
                 bf16_t* __restrict__ outB0, bf16_t* __restrict__ outB1,
                 bf16_t* __restrict__ outB2) {
  __shared__ bf16_t As[2][BM * LDA];
  __shared__ bf16_t Bs[2][BN * LDB];

  const int tid  = threadIdx.x;
  const int m0   = blockIdx.x * BM;
  const int n0   = blockIdx.y * BN;
  const int wid  = tid >> 5;
  const int lane = tid & 31;
  const int half = lane >> 4;
  const int lq   = lane & 15;
  const int wm   = (wid >> 1) * 32;   // 0,32,64,96
  const int wn   = (wid & 1) * 64;    // 0,64

  v8f acc[2][4];
#pragma unroll
  for (int mt = 0; mt < 2; mt++)
#pragma unroll
    for (int nt = 0; nt < 4; nt++)
#pragma unroll
      for (int e = 0; e < 8; e++) acc[mt][nt][e] = 0.0f;

  const int sr = tid >> 1;          // staged row 0..127
  const int sc = (tid & 1) * 16;    // staged col group
  const bf16_t* Arow = A + (size_t)(m0 + sr) * K + sc;
  const bf16_t* Brow = Bt + (size_t)(n0 + sr) * K + sc;

  // async-DMA one k-tile (4 b128 ops per thread: A lo/hi + B lo/hi)
  auto issue_tile = [&](int b, int ktile) {
    const bf16_t* ga = Arow + (size_t)ktile * BK;
    const bf16_t* gb = Brow + (size_t)ktile * BK;
    async_ld_b128(&As[b][sr * LDA + sc],     ga);
    async_ld_b128(&As[b][sr * LDA + sc + 8], ga + 8);
    async_ld_b128(&Bs[b][sr * LDB + sc],     gb);
    async_ld_b128(&Bs[b][sr * LDB + sc + 8], gb + 8);
  };

  issue_tile(0, 0);
  const int nk = K / BK;
  for (int kt = 0; kt < nk; kt++) {
    const int cur = kt & 1;
    const bool more = (kt + 1) < nk;
    if (more) {
      issue_tile(cur ^ 1, kt + 1);   // overlaps this tile's WMMAs
      wait_asynccnt<4>();            // oldest 4 = tile kt complete (in-order)
    } else {
      wait_asynccnt<0>();
    }
    __syncthreads();                 // all waves' DMA for tile kt visible

    // --- A fragments: 16x32, lane<16: K 0..7 & 16..23 ; lane>=16: K 8..15 & 24..31
    Frag af[2];
#pragma unroll
    for (int mt = 0; mt < 2; mt++) {
      const u32x4* p = reinterpret_cast<const u32x4*>(&As[cur][(wm + mt * 16 + lq) * LDA]);
      af[mt].q[0] = p[half];
      af[mt].q[1] = p[2 + half];
    }
    // --- B fragments: 32x16, lane<16: col n, K 0..7 & 16..23 ; lane>=16: K 8..15 & 24..31
    Frag bfr[4];
#pragma unroll
    for (int nt = 0; nt < 4; nt++) {
      const u32x4* p = reinterpret_cast<const u32x4*>(&Bs[cur][(wn + nt * 16 + lq) * LDB]);
      bfr[nt].q[0] = p[half];
      bfr[nt].q[1] = p[2 + half];
    }
#pragma unroll
    for (int mt = 0; mt < 2; mt++)
#pragma unroll
      for (int nt = 0; nt < 4; nt++)
        acc[mt][nt] = wmma_bf16(af[mt].v, bfr[nt].v, acc[mt][nt]);
    __syncthreads();                 // reads of buf cur done before it is re-filled
  }

  // --- epilogue: C layout -> row = e + 8*half, col = lq (per 16x16 tile) ---
  if constexpr (EPI == EPI_QKV) {
    // each 16x16 tile lies inside one (three, head, batch) region: hoist the scatter math
#pragma unroll
    for (int mt = 0; mt < 2; mt++) {
      const int rowb = m0 + wm + mt * 16;
      const int b    = rowb >> 11;
#pragma unroll
      for (int nt = 0; nt < 4; nt++) {
        const int colb  = n0 + wn + nt * 16;
        const int three = colb >> 10;
        const int h     = (colb & 1023) >> 6;
        const int d     = (colb & 63) + lq;
        const size_t bh = (size_t)(b * NHEAD + h);
#pragma unroll
        for (int e = 0; e < 8; e++) {
          const int l = (rowb + e + 8 * half) & 2047;
          const float val = acc[mt][nt][e];
          if (three == 0)      outB0[(bh * SEQ + l) * HDIM + d] = (bf16_t)val;  // Q [B,H,L,HD]
          else if (three == 1) outB1[(bh * SEQ + l) * HDIM + d] = (bf16_t)val;  // K [B,H,L,HD]
          else                 outB2[(bh * HDIM + d) * SEQ + l] = (bf16_t)val;  // V^T [B,H,HD,L]
        }
      }
    }
  } else {
#pragma unroll
    for (int mt = 0; mt < 2; mt++)
#pragma unroll
      for (int nt = 0; nt < 4; nt++)
#pragma unroll
        for (int e = 0; e < 8; e++) {
          const int row = m0 + wm + mt * 16 + e + 8 * half;
          const int col = n0 + wn + nt * 16 + lq;
          const float val = acc[mt][nt][e];
          if constexpr (EPI == EPI_RES) {
            const size_t idx = (size_t)row * N + col;
            outF[idx] = val + res[idx];
          } else if constexpr (EPI == EPI_GELU) {
            float t = val + bias[col];
            float g = 0.5f * t * (1.0f + erff(t * 0.70710678118654752f));  // exact GELU
            outB0[(size_t)row * N + col] = (bf16_t)g;
          } else {  // EPI_FFN2
            const size_t idx = (size_t)row * N + col;
            outF[idx] = val + bias[col] + res[idx];
          }
        }
  }
}

// ---------------- flash attention (causal) ----------------
// grid: (SEQ/64, NHEAD, NBATCH); block: 128 threads = 4 waves, 16 query rows each.
#define LDK 72
#define LDV 48
#define LDP 48
__global__ __launch_bounds__(128)
void k_flash_attn(const bf16_t* __restrict__ Q, const bf16_t* __restrict__ Kc,
                  const bf16_t* __restrict__ Vt, bf16_t* __restrict__ O) {
  __shared__ bf16_t Ks[32 * LDK];       // [key][d]
  __shared__ bf16_t Vs[HDIM * LDV];     // [d][key]
  __shared__ bf16_t Ps[4][16 * LDP];    // per-wave P re-swizzle scratch

  const int tid  = threadIdx.x;
  const int wid  = tid >> 5;
  const int lane = tid & 31;
  const int half = lane >> 4;
  const int lq   = lane & 15;
  const int qt   = blockIdx.x;
  const int h    = blockIdx.y;
  const int b    = blockIdx.z;
  const size_t head = (size_t)(b * NHEAD + h) * SEQ * HDIM;
  const int qbase = qt * 64 + wid * 16;

  // preload this wave's Q fragments (two 16x32 k-steps covering HD=64)
  Frag qf[2];
  {
    const u32x4* p = reinterpret_cast<const u32x4*>(Q + head + (size_t)(qbase + lq) * HDIM);
#pragma unroll
    for (int ks = 0; ks < 2; ks++) {
      qf[ks].q[0] = p[ks * 4 + half];
      qf[ks].q[1] = p[ks * 4 + 2 + half];
    }
  }

  v8f oacc[4];
  float mrun[8], lrun[8];
#pragma unroll
  for (int dj = 0; dj < 4; dj++)
#pragma unroll
    for (int e = 0; e < 8; e++) oacc[dj][e] = 0.0f;
#pragma unroll
  for (int e = 0; e < 8; e++) { mrun[e] = -3.0e38f; lrun[e] = 0.0f; }

  const int nkb = qt * 2 + 2;           // causal: key blocks of 32 up to q-tile end
  const int kr = tid >> 2, kp = (tid & 3) * 16;   // K staging: key, d-chunk
  const int vr = tid >> 1, vp = (tid & 1) * 16;   // V staging: d, key-chunk
  const bf16_t* Vhead = Vt + ((size_t)(b * NHEAD + h) * HDIM + vr) * SEQ + vp;

  for (int kbi = 0; kbi < nkb; kbi++) {
    const int k0 = kbi * 32;
    // async-DMA K tile [32 keys][64 d] and V^T tile [64 d][32 keys]
    const bf16_t* gk = Kc + head + (size_t)(k0 + kr) * HDIM + kp;
    async_ld_b128(&Ks[kr * LDK + kp],     gk);
    async_ld_b128(&Ks[kr * LDK + kp + 8], gk + 8);
    const bf16_t* gv = Vhead + k0;
    async_ld_b128(&Vs[vr * LDV + vp],     gv);
    async_ld_b128(&Vs[vr * LDV + vp + 8], gv + 8);
    wait_asynccnt<0>();
    __syncthreads();

    // S = Q * K^T  (two 16-key n-tiles)
    v8f s[2];
#pragma unroll
    for (int j = 0; j < 2; j++) {
#pragma unroll
      for (int e = 0; e < 8; e++) s[j][e] = 0.0f;
#pragma unroll
      for (int ks = 0; ks < 2; ks++) {
        Frag kf;
        const u32x4* p = reinterpret_cast<const u32x4*>(&Ks[(j * 16 + lq) * LDK]);
        kf.q[0] = p[ks * 4 + half];
        kf.q[1] = p[ks * 4 + 2 + half];
        s[j] = wmma_bf16(qf[ks].v, kf.v, s[j]);
      }
    }

    // online softmax (row r = e + 8*half lives across 16 lanes of this half;
    // xor masks 1/2/4/8 stay inside the 16-lane group)
#pragma unroll
    for (int e = 0; e < 8; e++) {
      const int qrow = qbase + e + 8 * half;
      float s0 = s[0][e] * ATT_SCALE;
      float s1 = s[1][e] * ATT_SCALE;
      if (k0 + lq > qrow)      s0 = -3.0e38f;   // causal mask
      if (k0 + 16 + lq > qrow) s1 = -3.0e38f;
      float mx = fmaxf(s0, s1);
      mx = fmaxf(mx, swz_xor<1>(mx));
      mx = fmaxf(mx, swz_xor<2>(mx));
      mx = fmaxf(mx, swz_xor<4>(mx));
      mx = fmaxf(mx, swz_xor<8>(mx));
      const float mnew = fmaxf(mrun[e], mx);
      const float corr = __expf(mrun[e] - mnew);
      const float p0 = __expf(s0 - mnew);
      const float p1 = __expf(s1 - mnew);
      float rs = p0 + p1;
      rs += swz_xor<1>(rs);
      rs += swz_xor<2>(rs);
      rs += swz_xor<4>(rs);
      rs += swz_xor<8>(rs);
      lrun[e] = lrun[e] * corr + rs;
      mrun[e] = mnew;
#pragma unroll
      for (int dj = 0; dj < 4; dj++) oacc[dj][e] *= corr;
      const int m = e + 8 * half;
      Ps[wid][m * LDP + lq]      = (bf16_t)p0;   // C-layout -> A-layout via LDS
      Ps[wid][m * LDP + 16 + lq] = (bf16_t)p1;
    }

    // O += P * V  (P is 16x32 A fragment; V^T rows give B fragments)
    Frag pf;
    {
      const u32x4* p = reinterpret_cast<const u32x4*>(&Ps[wid][lq * LDP]);
      pf.q[0] = p[half];
      pf.q[1] = p[2 + half];
    }
#pragma unroll
    for (int dj = 0; dj < 4; dj++) {
      Frag vf;
      const u32x4* p = reinterpret_cast<const u32x4*>(&Vs[(dj * 16 + lq) * LDV]);
      vf.q[0] = p[half];
      vf.q[1] = p[2 + half];
      oacc[dj] = wmma_bf16(pf.v, vf.v, oacc[dj]);
    }
    __syncthreads();
  }

  // finalize: O row-major [B*L, D] bf16, col = h*64 + d
#pragma unroll
  for (int e = 0; e < 8; e++) {
    const float inv = 1.0f / lrun[e];
    const int qrow = qbase + e + 8 * half;
    const size_t orow = ((size_t)b * SEQ + qrow) * D_MODEL + h * HDIM;
#pragma unroll
    for (int dj = 0; dj < 4; dj++)
      O[orow + dj * 16 + lq] = (bf16_t)(oacc[dj][e] * inv);
  }
}

// ---------------- row LayerNorm (fp32 math), dual fp32/bf16 output ----------------
__global__ __launch_bounds__(256)
void k_layernorm(const float* __restrict__ in, const float* __restrict__ g,
                 const float* __restrict__ be, float* __restrict__ outF,
                 bf16_t* __restrict__ outB) {
  const int row = blockIdx.x;
  const float* xr = in + (size_t)row * D_MODEL;
  float v[4], s = 0.0f, s2 = 0.0f;
#pragma unroll
  for (int i = 0; i < 4; i++) {
    float t = xr[threadIdx.x + 256 * i];
    v[i] = t; s += t; s2 += t * t;
  }
  s  += swz_xor<1>(s);   s2 += swz_xor<1>(s2);
  s  += swz_xor<2>(s);   s2 += swz_xor<2>(s2);
  s  += swz_xor<4>(s);   s2 += swz_xor<4>(s2);
  s  += swz_xor<8>(s);   s2 += swz_xor<8>(s2);
  s  += swz_xor<16>(s);  s2 += swz_xor<16>(s2);
  __shared__ float ws[8], ws2[8];
  const int wid = threadIdx.x >> 5, lane = threadIdx.x & 31;
  if (lane == 0) { ws[wid] = s; ws2[wid] = s2; }
  __syncthreads();
  s = 0.0f; s2 = 0.0f;
#pragma unroll
  for (int i = 0; i < 8; i++) { s += ws[i]; s2 += ws2[i]; }
  const float mu = s * (1.0f / D_MODEL);
  const float var = s2 * (1.0f / D_MODEL) - mu * mu;
  const float rstd = rsqrtf(var + 1e-5f);
#pragma unroll
  for (int i = 0; i < 4; i++) {
    const int c = threadIdx.x + 256 * i;
    const float y = (v[i] - mu) * rstd * g[c] + be[c];
    outF[(size_t)row * D_MODEL + c] = y;
    if (outB) outB[(size_t)row * D_MODEL + c] = (bf16_t)y;
  }
}

// ---------------- host launch ----------------
extern "C" void kernel_launch(void* const* d_in, const int* in_sizes, int n_in,
                              void* d_out, int out_size, void* d_ws, size_t ws_size,
                              hipStream_t stream) {
  const float* x     = (const float*)d_in[0];
  const float* w_qkv = (const float*)d_in[1];
  const float* w_out = (const float*)d_in[2];
  const float* w1    = (const float*)d_in[3];
  const float* b1    = (const float*)d_in[4];
  const float* w2    = (const float*)d_in[5];
  const float* b2    = (const float*)d_in[6];
  const float* g1    = (const float*)d_in[7];
  const float* be1   = (const float*)d_in[8];
  const float* g2    = (const float*)d_in[9];
  const float* be2   = (const float*)d_in[10];
  float* out = (float*)d_out;

  // bump-allocate scratch
  char* ws = (char*)d_ws;
  size_t off = 0;
  auto alloc = [&](size_t bytes) -> char* {
    char* p = ws + off;
    off += (bytes + 255) & ~(size_t)255;
    return p;
  };
  bf16_t* wqkvT  = (bf16_t*)alloc((size_t)3072 * 1024 * 2);
  bf16_t* woutT  = (bf16_t*)alloc((size_t)1024 * 1024 * 2);
  bf16_t* w1T    = (bf16_t*)alloc((size_t)2048 * 1024 * 2);
  bf16_t* w2T    = (bf16_t*)alloc((size_t)1024 * 2048 * 2);
  bf16_t* xb     = (bf16_t*)alloc((size_t)MTOK * 1024 * 2);
  bf16_t* qb     = (bf16_t*)alloc((size_t)MTOK * 1024 * 2);
  bf16_t* kb     = (bf16_t*)alloc((size_t)MTOK * 1024 * 2);
  bf16_t* vt     = (bf16_t*)alloc((size_t)MTOK * 1024 * 2);
  bf16_t* attnb  = (bf16_t*)alloc((size_t)MTOK * 1024 * 2);
  float*  res1   = (float*) alloc((size_t)MTOK * 1024 * 4);
  bf16_t* ln1b   = (bf16_t*)alloc((size_t)MTOK * 1024 * 2);
  float*  ln1f   = (float*) alloc((size_t)MTOK * 1024 * 4);
  bf16_t* h1     = (bf16_t*)alloc((size_t)MTOK * 2048 * 2);
  float*  tmp2   = (float*) alloc((size_t)MTOK * 1024 * 4);

  dim3 tb(32, 8);
  // weight transposes to bf16 [N,K]
  k_transpose_cvt<<<dim3(3072 / 32, 1024 / 32), tb, 0, stream>>>(w_qkv, wqkvT, 1024, 3072);
  k_transpose_cvt<<<dim3(1024 / 32, 1024 / 32), tb, 0, stream>>>(w_out, woutT, 1024, 1024);
  k_transpose_cvt<<<dim3(2048 / 32, 1024 / 32), tb, 0, stream>>>(w1,    w1T,   1024, 2048);
  k_transpose_cvt<<<dim3(1024 / 32, 2048 / 32), tb, 0, stream>>>(w2,    w2T,   2048, 1024);
  // x -> bf16
  const int nx = MTOK * D_MODEL;
  k_cvt_bf16<<<nx / 256, 256, 0, stream>>>(x, xb, nx);

  // QKV projection with head-scatter epilogue
  k_gemm_bf16<EPI_QKV><<<dim3(MTOK / 128, 3072 / 128), 256, 0, stream>>>(
      xb, wqkvT, MTOK, 3072, 1024, nullptr, nullptr, nullptr, qb, kb, vt);

  // causal flash attention
  k_flash_attn<<<dim3(SEQ / 64, NHEAD, NBATCH), 128, 0, stream>>>(qb, kb, vt, attnb);

  // output projection + residual(x)
  k_gemm_bf16<EPI_RES><<<dim3(MTOK / 128, 1024 / 128), 256, 0, stream>>>(
      attnb, woutT, MTOK, 1024, 1024, nullptr, x, res1, nullptr, nullptr, nullptr);

  // LN1 (fp32 residual copy + bf16 for next GEMM)
  k_layernorm<<<MTOK, 256, 0, stream>>>(res1, g1, be1, ln1f, ln1b);

  // FFN up + exact GELU
  k_gemm_bf16<EPI_GELU><<<dim3(MTOK / 128, 2048 / 128), 256, 0, stream>>>(
      ln1b, w1T, MTOK, 2048, 1024, b1, nullptr, nullptr, h1, nullptr, nullptr);

  // FFN down + bias + residual(ln1)
  k_gemm_bf16<EPI_FFN2><<<dim3(MTOK / 128, 1024 / 128), 256, 0, stream>>>(
      h1, w2T, MTOK, 1024, 2048, b2, ln1f, tmp2, nullptr, nullptr, nullptr);

  // LN2 -> final fp32 output
  k_layernorm<<<MTOK, 256, 0, stream>>>(tmp2, g2, be2, out, nullptr);
}
